// DCGRUCell_48026324304060
// MI455X (gfx1250) — matrix operand order repaired and checked by reading, not compiled, not run on hardware
//
#include <hip/hip_runtime.h>
#include <hip/hip_bf16.h>

typedef __attribute__((ext_vector_type(16))) __bf16 v16bf;
typedef __attribute__((ext_vector_type(8)))  __bf16 v8bf;
typedef __attribute__((ext_vector_type(4)))  __bf16 v4bf;
typedef __attribute__((ext_vector_type(8)))  float  v8f;

static constexpr int kN     = 4096;   // nodes
static constexpr int kB     = 64;     // batch
static constexpr int kUnits = 64;
static constexpr int kInDim = 2;
static constexpr int kInSz  = 66;     // kInDim + kUnits
static constexpr int kM     = 5;      // diffusion matrices
static constexpr int kF     = kInSz * kB;   // 4224 feature columns
static constexpr int kK     = kN;           // GEMM reduction dim
static constexpr int kProjK = kInSz * kM;   // 330

// Async global -> LDS copy (CDNA5, tracked by ASYNCcnt). lds32 is the low
// 32 bits of the flat shared address (== LDS byte offset per aperture spec).
#define ASYNC_LDS_B128(lds32, gptr)                                     \
  asm volatile("global_load_async_to_lds_b128 %0, %1, off"              \
               :: "v"(lds32), "v"(gptr) : "memory")

__device__ __forceinline__ unsigned lds_addr(const void* p) {
  return (unsigned)(unsigned long long)p;
}

// ---------------------------------------------------------------------------
// 1) f32 -> bf16 conversion for the supports (L2-resident afterwards)
// ---------------------------------------------------------------------------
__global__ void dcgru_cvt_bf16(const float* __restrict__ src,
                               __bf16* __restrict__ dst, int n4) {
  int i = blockIdx.x * blockDim.x + threadIdx.x;
  if (i >= n4) return;
  float4 v = ((const float4*)src)[i];
  v4bf o;
  o[0] = (__bf16)v.x; o[1] = (__bf16)v.y; o[2] = (__bf16)v.z; o[3] = (__bf16)v.w;
  ((v4bf*)dst)[i] = o;
}

// ---------------------------------------------------------------------------
// 2) Build Xt (kF x kN, bf16, k-contiguous) via LDS transpose.
//    Xt[(c*kB + b)][n] = c<2 ? inputs[b][n*2+c] : state[b][n*64 + c-2]
//    One block handles one batch b and 64 consecutive nodes: coalesced f32
//    reads -> padded LDS tile -> coalesced bf16 row writes.
// ---------------------------------------------------------------------------
__global__ void __launch_bounds__(256)
dcgru_build_xt(const float* __restrict__ inputs,
               const float* __restrict__ state,
               __bf16* __restrict__ xt) {
  __shared__ float st[64][65];   // [node][unit], pad -> conflict-free columns
  __shared__ float ti[64][2];    // [node][in_dim]

  const int tid = threadIdx.x;
  const int b   = blockIdx.x >> 6;          // 64 node-tiles per batch
  const int n0  = (blockIdx.x & 63) * 64;

  // Stage state tile: 64 nodes x 64 units, coalesced.
  for (int idx = tid; idx < 64 * 64; idx += 256) {
    int nr = idx >> 6, cc = idx & 63;
    st[nr][cc] = state[(size_t)b * (kN * kUnits) + (size_t)(n0 + nr) * kUnits + cc];
  }
  // Stage input tile: 64 nodes x 2.
  for (int idx = tid; idx < 64 * kInDim; idx += 256) {
    int nr = idx >> 1, cc = idx & 1;
    ti[nr][cc] = inputs[(size_t)b * (kN * kInDim) + (size_t)(n0 + nr) * kInDim + cc];
  }
  __syncthreads();

  // Write 66 feature rows of 64 contiguous bf16 each (coalesced).
  for (int idx = tid; idx < kInSz * 64; idx += 256) {
    int c = idx >> 6, ln = idx & 63;
    float v = (c < kInDim) ? ti[ln][c] : st[ln][c - kInDim];
    xt[(size_t)(c * kB + b) * kN + (n0 + ln)] = (__bf16)v;
  }
}

// ---------------------------------------------------------------------------
// 3) Diffusion GEMM:  Y^T = alpha * (S @ X)^T + beta * XC^T   (bf16 data,
//    f32 WMMA accumulation).  S: (4096,4096) row-major.  X, XC, Y stored
//    transposed: Mat[col][k].  Block tile 128x128, 8 waves (2x4), wave tile
//    64x32 -> 8 v_wmma per k-step.  Tiles staged with async global->LDS
//    copies; next tile's loads are in flight during current tile's WMMAs.
// ---------------------------------------------------------------------------
__global__ void __launch_bounds__(256)
dcgru_gemm_diffuse(const __bf16* __restrict__ S,
                   const __bf16* __restrict__ X,
                   const __bf16* __restrict__ XC,
                   __bf16* __restrict__ Y,
                   float alpha, float beta) {
  constexpr int TM = 128, TN = 128, TK = 32;
  constexpr int KT = kK / TK;          // 128 k-steps

  __shared__ __bf16 lsA[2][TM][TK];    // 8KB per buffer
  __shared__ __bf16 lsB[2][TN][TK];    // 8KB per buffer

  const int tid  = threadIdx.x;
  const int m0   = blockIdx.x * TM;    // S row tile
  const int n0   = blockIdx.y * TN;    // feature-column tile
  const int wave = tid >> 5;
  const int lane = tid & 31;
  const int hi   = lane >> 4;          // lane-half select
  const int l16  = lane & 15;
  const int wr   = wave >> 2;          // 0..1 (row group of 64)
  const int wc   = wave & 3;           // 0..3 (col group of 32)

  // Tile staging: 512 v8bf chunks per 128x32 tile, 2 per thread per matrix.
  const int c0 = tid,        rowG0 = c0 >> 2, kc0 = (c0 & 3) * 8;
  const int c1 = tid + 256,  rowG1 = c1 >> 2, kc1 = (c1 & 3) * 8;

  const __bf16* gA = S + (size_t)m0 * kK;
  const __bf16* gB = X + (size_t)n0 * kK;

  v8f acc[4][2];
  v8f zero = {0.f, 0.f, 0.f, 0.f, 0.f, 0.f, 0.f, 0.f};
  #pragma unroll
  for (int i = 0; i < 4; ++i)
    #pragma unroll
    for (int j = 0; j < 2; ++j) acc[i][j] = zero;

  auto issueTile = [&](int buf, int kt) {
    const __bf16* pA = gA + (size_t)kt * TK;
    const __bf16* pB = gB + (size_t)kt * TK;
    ASYNC_LDS_B128(lds_addr(&lsA[buf][rowG0][kc0]), pA + (size_t)rowG0 * kK + kc0);
    ASYNC_LDS_B128(lds_addr(&lsA[buf][rowG1][kc1]), pA + (size_t)rowG1 * kK + kc1);
    ASYNC_LDS_B128(lds_addr(&lsB[buf][rowG0][kc0]), pB + (size_t)rowG0 * kK + kc0);
    ASYNC_LDS_B128(lds_addr(&lsB[buf][rowG1][kc1]), pB + (size_t)rowG1 * kK + kc1);
  };

  issueTile(0, 0);                      // 4 async loads in flight

  for (int kt = 0; kt < KT; ++kt) {
    const int  buf  = kt & 1;
    const bool pref = (kt + 1 < KT);
    if (pref) {
      issueTile(buf ^ 1, kt + 1);       // overlap next tile with compute
      asm volatile("s_wait_asynccnt 4" ::: "memory");  // tile kt complete
    } else {
      asm volatile("s_wait_asynccnt 0" ::: "memory");
    }
    __syncthreads();                    // tile kt visible to all waves

    // A fragments (16x32 per row tile): lanes 0-15 hold K{0-7,16-23},
    // lanes 16-31 hold K{8-15,24-31} (ISA 7.12.2, 16-bit A layout).
    v16bf af[4];
    #pragma unroll
    for (int rt = 0; rt < 4; ++rt) {
      int row = wr * 64 + rt * 16 + l16;
      union { v16bf v; v8bf h[2]; } u;
      u.h[0] = *(const v8bf*)&lsA[buf][row][hi * 8];
      u.h[1] = *(const v8bf*)&lsA[buf][row][hi * 8 + 16];
      af[rt] = u.v;
    }
    // B fragments (32x16 per col tile): per-lane fixed column; lanes 0-15
    // hold K{0-7,16-23}, lanes 16-31 hold K{8-15,24-31} (matches A).
    v16bf bfv[2];
    #pragma unroll
    for (int ct = 0; ct < 2; ++ct) {
      int col = wc * 32 + ct * 16 + l16;
      union { v16bf v; v8bf h[2]; } u;
      u.h[0] = *(const v8bf*)&lsB[buf][col][hi * 8];
      u.h[1] = *(const v8bf*)&lsB[buf][col][hi * 8 + 16];
      bfv[ct] = u.v;
    }

    #pragma unroll
    for (int rt = 0; rt < 4; ++rt)
      #pragma unroll
      for (int ct = 0; ct < 2; ++ct)
        acc[rt][ct] = __builtin_amdgcn_wmma_f32_16x16x32_bf16(
            false, af[rt], false, bfv[ct], (short)0, acc[rt][ct], false, false);

    __syncthreads();                    // all waves done reading `buf`
  }

  // Epilogue: D layout -> per lane fixed N, 8 consecutive M. Writes are
  // 16B-contiguous in the transposed output (which is next GEMM's operand).
  #pragma unroll
  for (int rt = 0; rt < 4; ++rt) {
    #pragma unroll
    for (int ct = 0; ct < 2; ++ct) {
      int nloc = wc * 32 + ct * 16 + l16;
      int mloc = wr * 64 + rt * 16 + hi * 8;
      size_t off = (size_t)(n0 + nloc) * kK + (size_t)(m0 + mloc);
      v8f a = acc[rt][ct];
      v8bf o;
      if (beta != 0.0f) {
        v8bf xc = *(const v8bf*)(XC + off);
        #pragma unroll
        for (int e = 0; e < 8; ++e) o[e] = (__bf16)(alpha * a[e] + beta * (float)xc[e]);
      } else {
        #pragma unroll
        for (int e = 0; e < 8; ++e) o[e] = (__bf16)(alpha * a[e]);
      }
      *(v8bf*)(Y + off) = o;
    }
  }
}

// ---------------------------------------------------------------------------
// 4) Gate projection: gates = sigmoid(xcat @ W_fn + b_fn); emit r*h and u.
// ---------------------------------------------------------------------------
__global__ void __launch_bounds__(128)
dcgru_proj_gate(const __bf16* __restrict__ Xs,    // 5 mats, stride kF*kN halves
                const float* __restrict__ Wf,     // (330,128)
                const float* __restrict__ bfn,    // (128)
                const float* __restrict__ hx,
                float* __restrict__ rh,
                float* __restrict__ uo) {
  __shared__ float xrow[16][kProjK];
  const int    tid = threadIdx.x;
  const int    b   = blockIdx.x >> 8;            // 256 blocks per batch
  const int    n0  = (blockIdx.x & 255) * 16;
  const size_t XS  = (size_t)kF * kN;

  for (int idx = tid; idx < 16 * kProjK; idx += 128) {
    int r = idx / kProjK, k = idx - r * kProjK;
    int c = k / kM, m = k - c * kM;
    xrow[r][k] = (float)Xs[(size_t)m * XS + (size_t)(c * kB + b) * kN + (n0 + r)];
  }
  __syncthreads();

  const int j = tid;
  float acc[16];
  #pragma unroll
  for (int r = 0; r < 16; ++r) acc[r] = bfn[j];
  for (int k = 0; k < kProjK; ++k) {
    float wv = Wf[k * 128 + j];
    #pragma unroll
    for (int r = 0; r < 16; ++r) acc[r] = fmaf(xrow[r][k], wv, acc[r]);
  }
  #pragma unroll
  for (int r = 0; r < 16; ++r) {
    float sig = 1.0f / (1.0f + __expf(-acc[r]));
    int n = n0 + r;
    if (j < kUnits) {
      size_t o = (size_t)b * (kN * kUnits) + (size_t)n * kUnits + j;
      rh[o] = sig * hx[o];
    } else {
      size_t o = (size_t)b * (kN * kUnits) + (size_t)n * kUnits + (j - kUnits);
      uo[o] = sig;
    }
  }
}

// ---------------------------------------------------------------------------
// 5) Candidate projection + GRU output: c = tanh(xcat @ W_g + b_g);
//    out = u*h + (1-u)*c
// ---------------------------------------------------------------------------
__global__ void __launch_bounds__(128)
dcgru_proj_cand(const __bf16* __restrict__ Xs,
                const float* __restrict__ Wg,     // (330,64)
                const float* __restrict__ bg,     // (64)
                const float* __restrict__ hx,
                const float* __restrict__ uo,
                float* __restrict__ out) {
  __shared__ float xrow[16][kProjK];
  const int    tid = threadIdx.x;
  const int    b   = blockIdx.x >> 8;
  const int    n0  = (blockIdx.x & 255) * 16;
  const size_t XS  = (size_t)kF * kN;

  for (int idx = tid; idx < 16 * kProjK; idx += 128) {
    int r = idx / kProjK, k = idx - r * kProjK;
    int c = k / kM, m = k - c * kM;
    xrow[r][k] = (float)Xs[(size_t)m * XS + (size_t)(c * kB + b) * kN + (n0 + r)];
  }
  __syncthreads();

  const int j     = tid & 63;
  const int rbase = (tid >> 6) * 8;     // threads 0-63 -> rows 0-7, 64-127 -> 8-15
  float acc[8];
  #pragma unroll
  for (int r = 0; r < 8; ++r) acc[r] = bg[j];
  for (int k = 0; k < kProjK; ++k) {
    float wv = Wg[k * 64 + j];
    #pragma unroll
    for (int r = 0; r < 8; ++r) acc[r] = fmaf(xrow[rbase + r][k], wv, acc[r]);
  }
  #pragma unroll
  for (int r = 0; r < 8; ++r) {
    float c = tanhf(acc[r]);
    int n = n0 + rbase + r;
    size_t o = (size_t)b * (kN * kUnits) + (size_t)n * kUnits + j;
    float u = uo[o];
    out[o] = u * hx[o] + (1.0f - u) * c;
  }
}

// ---------------------------------------------------------------------------
// Orchestration
// ---------------------------------------------------------------------------
extern "C" void kernel_launch(void* const* d_in, const int* in_sizes, int n_in,
                              void* d_out, int out_size, void* d_ws, size_t ws_size,
                              hipStream_t stream) {
  const float* inputs = (const float*)d_in[0];
  const float* hx     = (const float*)d_in[1];
  const float* sup    = (const float*)d_in[2];
  const float* w_fn   = (const float*)d_in[3];
  const float* b_fn   = (const float*)d_in[4];
  const float* w_g    = (const float*)d_in[5];
  const float* b_g    = (const float*)d_in[6];
  float*       out    = (float*)d_out;

  // Workspace layout
  char*   ws   = (char*)d_ws;
  __bf16* Sbf  = (__bf16*)ws;                              // 2 * 4096*4096 bf16
  size_t  supB = (size_t)2 * kN * kN * sizeof(__bf16);     // 64 MiB
  size_t  xsB  = (size_t)kF * kN * sizeof(__bf16);         // 33 MiB per mat
  __bf16* Xs0  = (__bf16*)(ws + supB);
  __bf16* Xs1  = (__bf16*)(ws + supB + 1 * xsB);
  __bf16* Xs2  = (__bf16*)(ws + supB + 2 * xsB);
  __bf16* Xs3  = (__bf16*)(ws + supB + 3 * xsB);
  __bf16* Xs4  = (__bf16*)(ws + supB + 4 * xsB);
  size_t  gBy  = (size_t)kB * kN * kUnits * sizeof(float); // 64 MiB
  float*  rh   = (float*)(ws + supB + 5 * xsB);
  float*  uo   = (float*)(ws + supB + 5 * xsB + gBy);

  const __bf16* S0 = Sbf;
  const __bf16* S1 = Sbf + (size_t)kN * kN;

  // 1) supports -> bf16
  {
    int n4 = 2 * kN * kN / 4;
    dcgru_cvt_bf16<<<(n4 + 255) / 256, 256, 0, stream>>>(sup, Sbf, n4);
  }

  const dim3 gGemm(kN / 128, kF / 128);   // 32 x 33
  const int  nXtBlocks = kB * (kN / 64);  // 4096

  auto diffuse = [&](const float* state) {
    dcgru_build_xt<<<nXtBlocks, 256, 0, stream>>>(inputs, state, Xs0);
    // xs1 = S0 @ x0
    dcgru_gemm_diffuse<<<gGemm, 256, 0, stream>>>(S0, Xs0, nullptr, Xs1, 1.0f,  0.0f);
    // xs2 = 2*S0 @ xs1 - x0
    dcgru_gemm_diffuse<<<gGemm, 256, 0, stream>>>(S0, Xs1, Xs0,    Xs2, 2.0f, -1.0f);
    // xs3 = S1 @ xs1        (reference's x0c mutation: x0c == xs1 here)
    dcgru_gemm_diffuse<<<gGemm, 256, 0, stream>>>(S1, Xs1, nullptr, Xs3, 1.0f,  0.0f);
    // xs4 = 2*S1 @ xs3 - xs1
    dcgru_gemm_diffuse<<<gGemm, 256, 0, stream>>>(S1, Xs3, Xs1,    Xs4, 2.0f, -1.0f);
  };

  // gconv #1 (state = hx) -> gates
  diffuse(hx);
  dcgru_proj_gate<<<kB * (kN / 16), 128, 0, stream>>>(Xs0, w_fn, b_fn, hx, rh, uo);

  // gconv #2 (state = r*hx) -> candidate + output
  diffuse(rh);
  dcgru_proj_cand<<<kB * (kN / 16), 128, 0, stream>>>(Xs0, w_g, b_g, hx, uo, out);
}
